// TextTransformer_34643206209569
// MI455X (gfx1250) — matrix operand order repaired and checked by reading, not compiled, not run on hardware
//
#include <hip/hip_runtime.h>
#include <hip/hip_bf16.h>
#include <math.h>

#define D_MODEL 768
#define N_HEAD  12
#define D_HEAD  64
#define FFDIM   3072
#define NLAYER  4
#define BATCH   32
#define SEQ     512
#define MROWS   (BATCH*SEQ)

typedef __bf16 bf16;
typedef __attribute__((ext_vector_type(16))) __bf16 v16bf;
typedef __attribute__((ext_vector_type(8)))  float  v8f;

union FragB { v16bf v; uint4 u[2]; };

__device__ __forceinline__ v8f v8f_zero() {
  v8f z;
#pragma unroll
  for (int i = 0; i < 8; i++) z[i] = 0.0f;
  return z;
}

__device__ __forceinline__ bf16 f2bf(float f) { return (bf16)f; }

__device__ __forceinline__ v8f wmma_bf16(v16bf a, v16bf b, v8f c) {
  // D = A(16x32 bf16) * B(32x16 bf16) + C(16x16 f32)
  return __builtin_amdgcn_wmma_f32_16x16x32_bf16(false, a, false, b, (short)0, c, false, false);
}

// CDNA5 async DMA: global -> LDS, tracked by ASYNCcnt. INST_OFFSET applies to
// both the LDS and global addresses, so one (ldsAddr, gaddr) pair covers a
// 32-byte half-row with two b128 ops.
__device__ __forceinline__ void async_g2l_b128(unsigned ldsAddr, const void* g) {
  unsigned long long ga = (unsigned long long)(size_t)g;
  asm volatile("global_load_async_to_lds_b128 %0, %1, off"
               :: "v"(ldsAddr), "v"(ga) : "memory");
  asm volatile("global_load_async_to_lds_b128 %0, %1, off offset:16"
               :: "v"(ldsAddr), "v"(ga) : "memory");
}
__device__ __forceinline__ void wait_async0() {
  asm volatile("s_wait_asynccnt 0x0" ::: "memory");
}

// ---------------------------------------------------------------- conversions
__global__ __launch_bounds__(256) void cvt_f32_bf16(const float* __restrict__ s,
                                                    bf16* __restrict__ d, int n) {
  int i = blockIdx.x * 256 + threadIdx.x;
  if (i < n) d[i] = f2bf(s[i]);
}

// ------------------------------------------------------------ embedding + PE
__global__ __launch_bounds__(256) void embed_kernel(const int* __restrict__ tokens,
                                                    const float* __restrict__ emb,
                                                    float* __restrict__ xf,
                                                    bf16* __restrict__ xb) {
  const int row = blockIdx.x;          // b*SEQ + s
  const int s   = row & (SEQ - 1);
  const int tok = tokens[row];
  const long base = (long)row * D_MODEL;
  const long ebase = (long)tok * D_MODEL;
#pragma unroll
  for (int i = 0; i < 3; i++) {
    int d = threadIdx.x + 256 * i;
    float e = (2.0f * (float)(d >> 1)) * (1.0f / 768.0f);
    float ang = (float)s * __expf(-9.210340371976184f * e); // 10000^-e
    float pe = (d & 1) ? cosf(ang) : sinf(ang);
    float v = emb[ebase + d] * 27.712812921102035f + pe;    // sqrt(768)
    xf[base + d] = v;
    xb[base + d] = f2bf(v);
  }
}

// ------------------------------------------------------------------ WMMA GEMM
// C[M,N] = act(A[M,K] * W[K,N] + bias); A,W bf16 row-major; out f32 and/or bf16
#define BM 128
#define BN 64
#define BK 32
#define LDP 40   // padded LDS stride (bf16 elems): 80B rows, 16B aligned, bank-spread

__global__ __launch_bounds__(256) void gemm_bias(const bf16* __restrict__ A,
                                                 const bf16* __restrict__ W,
                                                 const float* __restrict__ bias,
                                                 float* __restrict__ Cf,
                                                 bf16* __restrict__ Cb,
                                                 int M, int N, int K, int relu) {
  __shared__ bf16 As[BM * LDP];
  __shared__ bf16 Bs[BN * LDP];  // stored transposed: Bs[n][k]
  const int tid   = threadIdx.x;
  const int wave  = tid >> 5, lane = tid & 31;
  const int lrow  = lane & 15, lhalf = lane >> 4;
  const int kb8   = lhalf * 8;
  const int wm    = wave >> 1, wn = wave & 1;
  const long row0 = (long)blockIdx.y * BM;
  const long col0 = (long)blockIdx.x * BN;

  v8f acc[2][2];
#pragma unroll
  for (int a = 0; a < 2; a++)
#pragma unroll
    for (int b = 0; b < 2; b++) acc[a][b] = v8f_zero();

  // A staging map: each thread owns one 32B half-row of the 128x32 tile,
  // copied global->LDS by the CDNA5 async DMA path (no VGPR round-trip).
  const int arow = tid >> 1;
  const int acol = (tid & 1) * 16;
  const unsigned asAddr = (unsigned)(size_t)(&As[arow * LDP + acol]);
  // B staging map: each thread loads one 8-elem (16B) chunk of one W row,
  // stores transposed into Bs. Branch-free, fully vectorized global load.
  const int bkk = tid >> 3;        // 0..31 : k row within tile
  const int bn8 = (tid & 7) * 8;   // 0..56 : col group

  // hoisted, stride-advanced global pointers (no per-iter 64-bit mul)
  const bf16* aptr = A + (row0 + arow) * (long)K + acol;
  const bf16* wptr = W + (long)bkk * N + col0 + bn8;
  const long  wstep = (long)BK * N;

  const int kTiles = K / BK;
  for (int kt = 0; kt < kTiles; kt++) {
    __syncthreads();                 // fragments of previous tile consumed
    // stage A: two async b128 DMA ops into padded LDS
    async_g2l_b128(asAddr, aptr);
    // stage B transposed (one 16B load, 8 b16 LDS stores)
    {
      uint4 wv = *(const uint4*)wptr;
      union { uint4 u; bf16 h[8]; } t; t.u = wv;
#pragma unroll
      for (int e = 0; e < 8; e++) Bs[(bn8 + e) * LDP + bkk] = t.h[e];
    }
    if (kt + 1 < kTiles) {
      __builtin_prefetch(aptr + BK, 0, 1);
      __builtin_prefetch(wptr + wstep, 0, 1);
    }
    aptr += BK;
    wptr += wstep;
    wait_async0();                   // async A copies landed in LDS
    __syncthreads();

    FragB af[2], bfr[2];
#pragma unroll
    for (int tm = 0; tm < 2; tm++) {
      int r = wm * 32 + tm * 16 + lrow;
      af[tm].u[0] = *(const uint4*)(&As[r * LDP + kb8]);
      af[tm].u[1] = *(const uint4*)(&As[r * LDP + 16 + kb8]);
    }
#pragma unroll
    for (int tn = 0; tn < 2; tn++) {
      int c = wn * 32 + tn * 16 + lrow;
      bfr[tn].u[0] = *(const uint4*)(&Bs[c * LDP + kb8]);
      bfr[tn].u[1] = *(const uint4*)(&Bs[c * LDP + 16 + kb8]);
    }
#pragma unroll
    for (int tm = 0; tm < 2; tm++)
#pragma unroll
      for (int tn = 0; tn < 2; tn++)
        acc[tm][tn] = wmma_bf16(af[tm].v, bfr[tn].v, acc[tm][tn]);
  }

#pragma unroll
  for (int tm = 0; tm < 2; tm++) {
#pragma unroll
    for (int tn = 0; tn < 2; tn++) {
      long gcol = col0 + wn * 32 + tn * 16 + lrow;
      float bv = bias[gcol];
#pragma unroll
      for (int r = 0; r < 8; r++) {
        long grow = row0 + wm * 32 + tm * 16 + lhalf * 8 + r;
        float v = acc[tm][tn][r] + bv;
        if (relu) v = fmaxf(v, 0.0f);
        if (Cf) Cf[grow * N + gcol] = v;
        if (Cb) Cb[grow * N + gcol] = f2bf(v);
      }
    }
  }
}

// ----------------------------------------------------- fused flash attention
#define QBLK 128
#define SKT  32
#define PST  40
#define VST  40

__global__ __launch_bounds__(256) void attn_kernel(const bf16* __restrict__ Q,
                                                   const bf16* __restrict__ Km,
                                                   const bf16* __restrict__ V,
                                                   bf16* __restrict__ O) {
  __shared__ bf16 Pbuf[8 * 16 * PST];     // per-wave P tiles (16x32 bf16)
  __shared__ bf16 Vt[D_HEAD * VST];       // V tile transposed: Vt[dh][sk]
  const int bh = blockIdx.x;
  const int b = bh / N_HEAD, h = bh % N_HEAD;
  const int q0 = blockIdx.y * QBLK;
  const int tid = threadIdx.x;
  const int wave = tid >> 5, lane = tid & 31;
  const int lrow = lane & 15, lhalf = lane >> 4;
  const int kb8 = lhalf * 8;

  const long baseBH = (long)b * SEQ * D_MODEL + h * D_HEAD;

  // Q A-fragments: 16 rows x 64 dh -> two K=32 chunks
  FragB qf[2];
  {
    const bf16* qp = Q + baseBH + (long)(q0 + wave * 16 + lrow) * D_MODEL;
    qf[0].u[0] = *(const uint4*)(qp + kb8);
    qf[0].u[1] = *(const uint4*)(qp + 16 + kb8);
    qf[1].u[0] = *(const uint4*)(qp + 32 + kb8);
    qf[1].u[1] = *(const uint4*)(qp + 48 + kb8);
  }

  v8f oacc[4];
#pragma unroll
  for (int t = 0; t < 4; t++) oacc[t] = v8f_zero();
  float mrow[8], lsum[8];
#pragma unroll
  for (int r = 0; r < 8; r++) { mrow[r] = -1e30f; lsum[r] = 0.0f; }

  bf16* Pw = Pbuf + wave * 16 * PST;
  const int vsk = tid >> 3;
  const int vdh = (tid & 7) * 8;

  // hoisted, stride-advanced pointers
  const bf16* vp  = V  + baseBH + (long)vsk * D_MODEL + vdh;
  const bf16* kp  = Km + baseBH + (long)lrow * D_MODEL + lhalf * 8;
  const long  step = (long)SKT * D_MODEL;
  const long  krow16 = (long)16 * D_MODEL;

  for (int sk0 = 0; sk0 < SEQ; sk0 += SKT) {
    __syncthreads();   // protect Vt WAR from previous iteration
    {  // stage V tile (32 x 64) transposed into LDS (one 16B load, 8 b16 stores)
      uint4 vv = *(const uint4*)vp;
      union { uint4 u; bf16 h[8]; } t; t.u = vv;
#pragma unroll
      for (int e = 0; e < 8; e++) Vt[(vdh + e) * VST + vsk] = t.h[e];
    }
    __syncthreads();

    // scores: Q(16x64) * K^T(64x32) via WMMA, K B-fragments straight from global
    v8f s[2];
#pragma unroll
    for (int j = 0; j < 2; j++) {
      s[j] = v8f_zero();
      const bf16* kpj = kp + j * krow16;
      FragB kf0, kf1;
      kf0.u[0] = *(const uint4*)(kpj);
      kf0.u[1] = *(const uint4*)(kpj + 16);
      kf1.u[0] = *(const uint4*)(kpj + 32);
      kf1.u[1] = *(const uint4*)(kpj + 48);
      s[j] = wmma_bf16(qf[0].v, kf0.v, s[j]);
      s[j] = wmma_bf16(qf[1].v, kf1.v, s[j]);
    }
    vp += step;
    kp += step;

    // online softmax over the 32 new columns; write P (bf16) to per-wave LDS
#pragma unroll
    for (int r = 0; r < 8; r++) {
      float s0 = s[0][r] * 0.125f;   // 1/sqrt(64)
      float s1 = s[1][r] * 0.125f;
      float mx = fmaxf(s0, s1);
      mx = fmaxf(mx, __shfl_xor(mx, 1, 32));
      mx = fmaxf(mx, __shfl_xor(mx, 2, 32));
      mx = fmaxf(mx, __shfl_xor(mx, 4, 32));
      mx = fmaxf(mx, __shfl_xor(mx, 8, 32));
      float mnew  = fmaxf(mrow[r], mx);
      float alpha = __expf(mrow[r] - mnew);
      float p0 = __expf(s0 - mnew);
      float p1 = __expf(s1 - mnew);
      float rs = p0 + p1;
      rs += __shfl_xor(rs, 1, 32);
      rs += __shfl_xor(rs, 2, 32);
      rs += __shfl_xor(rs, 4, 32);
      rs += __shfl_xor(rs, 8, 32);
      lsum[r] = lsum[r] * alpha + rs;
      mrow[r] = mnew;
#pragma unroll
      for (int t = 0; t < 4; t++) oacc[t][r] *= alpha;
      int prow = lhalf * 8 + r;          // C-layout row
      Pw[prow * PST + lrow]      = f2bf(p0);
      Pw[prow * PST + 16 + lrow] = f2bf(p1);
    }
    __builtin_amdgcn_wave_barrier();  // LDS is in-order per wave; pin scheduling

    // P as A-fragment (16x32), V as B-fragments from transposed LDS tile
    FragB pf;
    pf.u[0] = *(const uint4*)(&Pw[lrow * PST + kb8]);
    pf.u[1] = *(const uint4*)(&Pw[lrow * PST + 16 + kb8]);
#pragma unroll
    for (int t = 0; t < 4; t++) {
      FragB vf;
      vf.u[0] = *(const uint4*)(&Vt[(t * 16 + lrow) * VST + kb8]);
      vf.u[1] = *(const uint4*)(&Vt[(t * 16 + lrow) * VST + 16 + kb8]);
      oacc[t] = wmma_bf16(pf.v, vf.v, oacc[t]);
    }
  }

  // normalize and store attention output (bf16, feeds O-projection GEMM)
#pragma unroll
  for (int t = 0; t < 4; t++) {
#pragma unroll
    for (int r = 0; r < 8; r++) {
      float inv = 1.0f / lsum[r];
      long srow = q0 + wave * 16 + lhalf * 8 + r;
      O[baseBH + srow * D_MODEL + t * 16 + lrow] = f2bf(oacc[t][r] * inv);
    }
  }
}

// ------------------------------------------------------- residual + layernorm
__global__ __launch_bounds__(256) void add_ln_kernel(const float* __restrict__ X,
                                                     const float* __restrict__ T,
                                                     const float* __restrict__ G,
                                                     const float* __restrict__ Bt,
                                                     float* __restrict__ Of,
                                                     bf16* __restrict__ Ob) {
  __shared__ float red[256];
  const int row = blockIdx.x;
  const long base = (long)row * D_MODEL;
  const int tid = threadIdx.x;
  float v[3];
  float s = 0.0f;
#pragma unroll
  for (int i = 0; i < 3; i++) {
    int c = tid + 256 * i;
    v[i] = X[base + c] + T[base + c];
    s += v[i];
  }
  red[tid] = s; __syncthreads();
  for (int o = 128; o > 0; o >>= 1) { if (tid < o) red[tid] += red[tid + o]; __syncthreads(); }
  float mean = red[0] * (1.0f / 768.0f);
  __syncthreads();
  float s2 = 0.0f;
#pragma unroll
  for (int i = 0; i < 3; i++) { float d = v[i] - mean; s2 += d * d; }
  red[tid] = s2; __syncthreads();
  for (int o = 128; o > 0; o >>= 1) { if (tid < o) red[tid] += red[tid + o]; __syncthreads(); }
  float inv = rsqrtf(red[0] * (1.0f / 768.0f) + 1e-6f);
#pragma unroll
  for (int i = 0; i < 3; i++) {
    int c = tid + 256 * i;
    float y = G[c] * (v[i] - mean) * inv + Bt[c];
    Of[base + c] = y;
    Ob[base + c] = f2bf(y);
  }
}

// -------------------------------------------------- pooling + GELU/sigmoid head
__global__ __launch_bounds__(256) void head_kernel(const float* __restrict__ X,
                                                   const float* __restrict__ wl,
                                                   const float* __restrict__ bl,
                                                   const float* __restrict__ wout,
                                                   const float* __restrict__ bout,
                                                   float* __restrict__ out) {
  __shared__ float pooled[D_MODEL];
  __shared__ float hsh[32];
  const int b = blockIdx.x;
#pragma unroll
  for (int i = 0; i < 3; i++) {
    int d = threadIdx.x + 256 * i;
    const float* p = X + (long)b * SEQ * D_MODEL + d;
    float s = 0.0f;
    for (int t = 0; t < SEQ; t++) s += p[(long)t * D_MODEL];
    pooled[d] = s * (1.0f / (float)SEQ);
  }
  __syncthreads();
  if (threadIdx.x < 32) {
    float a = bl[threadIdx.x];
    for (int d = 0; d < D_MODEL; d++) a += pooled[d] * wl[d * 32 + threadIdx.x];
    hsh[threadIdx.x] = 0.5f * a * (1.0f + erff(a * 0.70710678118654752f));
  }
  __syncthreads();
  if (threadIdx.x == 0) {
    float a = bout[0];
    for (int j = 0; j < 32; j++) a += hsh[j] * wout[j];
    out[b] = 1.0f / (1.0f + __expf(-a));
  }
}

// --------------------------------------------------------------------- driver
extern "C" void kernel_launch(void* const* d_in, const int* in_sizes, int n_in,
                              void* d_out, int out_size, void* d_ws, size_t ws_size,
                              hipStream_t stream) {
  (void)in_sizes; (void)n_in; (void)out_size; (void)ws_size;
  const int*   tokens = (const int*)  d_in[0];
  const float* emb    = (const float*)d_in[1];
  const float* wq = (const float*)d_in[2];
  const float* bq = (const float*)d_in[3];
  const float* wk = (const float*)d_in[4];
  const float* bk = (const float*)d_in[5];
  const float* wv = (const float*)d_in[6];
  const float* bv = (const float*)d_in[7];
  const float* wo = (const float*)d_in[8];
  const float* bo = (const float*)d_in[9];
  const float* w1 = (const float*)d_in[10];
  const float* b1 = (const float*)d_in[11];
  const float* w2 = (const float*)d_in[12];
  const float* b2 = (const float*)d_in[13];
  const float* ln1g = (const float*)d_in[14];
  const float* ln1b = (const float*)d_in[15];
  const float* ln2g = (const float*)d_in[16];
  const float* ln2b = (const float*)d_in[17];
  const float* wl   = (const float*)d_in[18];
  const float* bl   = (const float*)d_in[19];
  const float* wout = (const float*)d_in[20];
  const float* bout = (const float*)d_in[21];
  float* out = (float*)d_out;

  char* p = (char*)d_ws;
  auto take = [&](size_t bytes) -> char* {
    char* r = p; p += (bytes + 255) & ~(size_t)255; return r;
  };
  float* xf  = (float*)take((size_t)MROWS * D_MODEL * 4);
  float* t1f = (float*)take((size_t)MROWS * D_MODEL * 4);
  float* o1f = (float*)take((size_t)MROWS * D_MODEL * 4);
  bf16*  ab  = (bf16*) take((size_t)MROWS * D_MODEL * 2);
  bf16*  qb  = (bf16*) take((size_t)MROWS * D_MODEL * 2);
  bf16*  kbf = (bf16*) take((size_t)MROWS * D_MODEL * 2);
  bf16*  vbf = (bf16*) take((size_t)MROWS * D_MODEL * 2);
  bf16*  avb = (bf16*) take((size_t)MROWS * D_MODEL * 2);
  bf16*  hb  = (bf16*) take((size_t)MROWS * FFDIM   * 2);
  bf16*  wqb = (bf16*) take((size_t)NLAYER * D_MODEL * D_MODEL * 2);
  bf16*  wkb = (bf16*) take((size_t)NLAYER * D_MODEL * D_MODEL * 2);
  bf16*  wvb = (bf16*) take((size_t)NLAYER * D_MODEL * D_MODEL * 2);
  bf16*  wob = (bf16*) take((size_t)NLAYER * D_MODEL * D_MODEL * 2);
  bf16*  w1b = (bf16*) take((size_t)NLAYER * D_MODEL * FFDIM * 2);
  bf16*  w2b = (bf16*) take((size_t)NLAYER * FFDIM * D_MODEL * 2);

  // one-time (per launch) bf16 weight conversion
  const int nw = NLAYER * D_MODEL * D_MODEL;
  const int nf = NLAYER * D_MODEL * FFDIM;
  cvt_f32_bf16<<<(nw + 255) / 256, 256, 0, stream>>>(wq, wqb, nw);
  cvt_f32_bf16<<<(nw + 255) / 256, 256, 0, stream>>>(wk, wkb, nw);
  cvt_f32_bf16<<<(nw + 255) / 256, 256, 0, stream>>>(wv, wvb, nw);
  cvt_f32_bf16<<<(nw + 255) / 256, 256, 0, stream>>>(wo, wob, nw);
  cvt_f32_bf16<<<(nf + 255) / 256, 256, 0, stream>>>(w1, w1b, nf);
  cvt_f32_bf16<<<(nf + 255) / 256, 256, 0, stream>>>(w2, w2b, nf);

  embed_kernel<<<MROWS, 256, 0, stream>>>(tokens, emb, xf, ab);

  const dim3 gP (D_MODEL / BN, MROWS / BM);   // (12, 128)
  const dim3 gF1(FFDIM   / BN, MROWS / BM);   // (48, 128)
  const dim3 gA (BATCH * N_HEAD, SEQ / QBLK); // (384, 4)

  for (int l = 0; l < NLAYER; l++) {
    const bf16* Wq = wqb + (size_t)l * D_MODEL * D_MODEL;
    const bf16* Wk = wkb + (size_t)l * D_MODEL * D_MODEL;
    const bf16* Wv = wvb + (size_t)l * D_MODEL * D_MODEL;
    const bf16* Wo = wob + (size_t)l * D_MODEL * D_MODEL;
    const bf16* W1 = w1b + (size_t)l * D_MODEL * FFDIM;
    const bf16* W2 = w2b + (size_t)l * FFDIM * D_MODEL;

    gemm_bias<<<gP, 256, 0, stream>>>(ab, Wq, bq + l * D_MODEL, nullptr, qb,
                                      MROWS, D_MODEL, D_MODEL, 0);
    gemm_bias<<<gP, 256, 0, stream>>>(ab, Wk, bk + l * D_MODEL, nullptr, kbf,
                                      MROWS, D_MODEL, D_MODEL, 0);
    gemm_bias<<<gP, 256, 0, stream>>>(ab, Wv, bv + l * D_MODEL, nullptr, vbf,
                                      MROWS, D_MODEL, D_MODEL, 0);
    attn_kernel<<<gA, 256, 0, stream>>>(qb, kbf, vbf, avb);
    gemm_bias<<<gP, 256, 0, stream>>>(avb, Wo, bo + l * D_MODEL, t1f, nullptr,
                                      MROWS, D_MODEL, D_MODEL, 0);
    add_ln_kernel<<<MROWS, 256, 0, stream>>>(xf, t1f, ln1g + l * D_MODEL,
                                             ln1b + l * D_MODEL, o1f, ab);
    gemm_bias<<<gF1, 256, 0, stream>>>(ab, W1, b1 + l * FFDIM, nullptr, hb,
                                       MROWS, FFDIM, D_MODEL, 1);
    gemm_bias<<<gP, 256, 0, stream>>>(hb, W2, b2 + l * D_MODEL, t1f, nullptr,
                                      MROWS, D_MODEL, FFDIM, 0);
    add_ln_kernel<<<MROWS, 256, 0, stream>>>(o1f, t1f, ln2g + l * D_MODEL,
                                             ln2b + l * D_MODEL, xf, ab);
  }

  head_kernel<<<BATCH, 256, 0, stream>>>(xf, wl, bl, wout, bout, out);
}
